// SDP16_30021821399364
// MI455X (gfx1250) — compile-verified, matrix operand order checked by reading, not run on hardware
//
#include <hip/hip_runtime.h>
#include <hip/hip_bf16.h>

typedef __attribute__((ext_vector_type(16))) _Float16 v16h;
typedef __attribute__((ext_vector_type(8)))  float    v8f;

#define E32 32
#define GQ  2

union H8 { uint4 u; _Float16 h[8]; };

__device__ __forceinline__ int lane_id() { return (int)(threadIdx.x & 31); }

__device__ __forceinline__ v8f wmma_f16(v16h a, v16h b, v8f c) {
  return __builtin_amdgcn_wmma_f32_16x16x32_f16(
      /*neg_a=*/false, a, /*neg_b=*/false, b,
      /*c_mod=*/(short)0, c, /*reuse_a=*/false, /*reuse_b=*/false);
}

// Load a 16x32 fragment in the documented 16-bit A layout from a row-major
// source (16 rows x >=32 cols, row pitch = stride elements). For the B
// operand, pass B^T stored row-major (i.e. source[n][k]).
// Lane l: row=l&15, kb=8*(l>>4); halfs[0..7]=K kb..kb+7, halfs[8..15]=K kb+16..kb+23.
template <typename T>
__device__ __forceinline__ v16h load_frag(const T* __restrict__ src, int stride) {
  const int l  = lane_id();
  const int r  = l & 15;
  const int kb = (l >> 4) << 3;
  const T* p = src + (size_t)r * stride + kb;
  v16h f;
#pragma unroll
  for (int i = 0; i < 8; ++i) {
    f[i]     = (_Float16)p[i];
    f[8 + i] = (_Float16)p[i + 16];
  }
  return f;
}

// Copy a 16x32 f16 LDS tile to a fully-contiguous global region: 2 x b128/lane.
__device__ __forceinline__ void store_tile_f16(_Float16* __restrict__ dst,
                                               const _Float16* __restrict__ lds) {
  const int l = lane_id();
#pragma unroll
  for (int i = 0; i < 2; ++i) {
    const int idx = i * 32 + l;                    // uint4 index (64 total)
    *(uint4*)(dst + idx * 8) = *(const uint4*)(lds + idx * 8);
  }
}

// ---------------- Kernel 1: grouped QKV projection ----------------
// x: (B*N, 64) f32.  w: (192, 32) f32.  bias: (192,) f32.
// q/k/v f16 in (b,g,n,e) layout; vT f16 in (b,g,e,n) layout for the PV gemm.
__global__ void __launch_bounds__(32)
qkv_kernel(const float* __restrict__ x, const float* __restrict__ w,
           const float* __restrict__ bias,
           _Float16* __restrict__ q, _Float16* __restrict__ k,
           _Float16* __restrict__ v, _Float16* __restrict__ vT) {
  __shared__ alignas(16) _Float16 tile[16 * 32];

  const int t  = blockIdx.x;            // 16-row tile over B*N = 32768
  const int g  = blockIdx.y;
  const int l  = lane_id();
  const int rowbase = t * 16;
  const int b     = rowbase >> 10;      // tiles never cross batch boundary
  const int nbase = rowbase & 1023;
  const int bg    = b * GQ + g;
  const int nn = l & 15, half = l >> 4;

  v16h a = load_frag(x + (size_t)rowbase * (GQ * E32) + g * E32, GQ * E32);

  _Float16* outs[3] = {q, k, v};
#pragma unroll
  for (int s = 0; s < 3; ++s) {         // q, k, v
    v8f acc[2];
    float bn[2];
#pragma unroll
    for (int cb = 0; cb < 2; ++cb) {
      const int obase = g * 96 + s * 32 + cb * 16;
      v16h bf = load_frag(w + (size_t)obase * E32, E32);
      v8f z = {};
      acc[cb] = wmma_f16(a, bf, z);
      bn[cb]  = bias[obase + nn];
    }
    __syncthreads();                    // protect tile from previous read
#pragma unroll
    for (int cb = 0; cb < 2; ++cb) {
#pragma unroll
      for (int r = 0; r < 8; ++r) {
        const int m = r + half * 8;
        tile[m * 32 + cb * 16 + nn] = (_Float16)(acc[cb][r] + bn[cb]);
      }
    }
    if (s == 2) {                       // also write V^T, packed b128 per block
#pragma unroll
      for (int cb = 0; cb < 2; ++cb) {
        H8 p;
#pragma unroll
        for (int r = 0; r < 8; ++r) p.h[r] = (_Float16)(acc[cb][r] + bn[cb]);
        const int e = cb * 16 + nn;     // lane's 8 rows are consecutive keys
        *(uint4*)(vT + ((size_t)bg * E32 + e) * 1024 + nbase + half * 8) = p.u;
      }
    }
    __syncthreads();
    store_tile_f16(outs[s] + ((size_t)bg * 1024 + nbase) * E32, tile);
  }
}

// ---------------- Kernel 2: flash attention per (b,g) ----------------
// 16 query rows per wave; 64 keys per step (4 score WMMAs + 4 PV WMMAs).
__global__ void __launch_bounds__(32)
attn_kernel(const _Float16* __restrict__ q, const _Float16* __restrict__ k,
            const _Float16* __restrict__ vT, _Float16* __restrict__ attn) {
  __shared__ alignas(16) _Float16 pTile[16 * 64];

  const int qt = blockIdx.x;            // query tile 0..63
  const int bg = blockIdx.y;            // (b*G+g) 0..63
  const int l  = lane_id();
  const _Float16* kb  = k  + (size_t)bg * 1024 * E32;
  const _Float16* vtb = vT + (size_t)bg * E32 * 1024;

  v16h aq = load_frag(q + ((size_t)bg * 1024 + qt * 16) * E32, E32);

  float mrow[8], lrow[8];
  v8f acc0 = {}, acc1 = {};
#pragma unroll
  for (int r = 0; r < 8; ++r) { mrow[r] = -1e30f; lrow[r] = 0.0f; }

  const float scale = 1.0f / (float)E32;
  const int nn = l & 15, half = l >> 4;

  for (int jt = 0; jt < 16; ++jt) {     // 16 chunks x 64 keys = 1024
    const int jb = jt * 64;
    if (jt < 15) {                      // prefetch next K / V^T chunk
      __builtin_prefetch(kb + (size_t)(jb + 64) * E32, 0, 1);
      __builtin_prefetch(vtb + jb + 64, 0, 1);
    }
    v8f s[4];
#pragma unroll
    for (int c = 0; c < 4; ++c) {
      v16h bk = load_frag(kb + (size_t)(jb + c * 16) * E32, E32);
      v8f z = {};
      s[c] = wmma_f16(aq, bk, z);
    }

    float p[4][8];
#pragma unroll
    for (int r = 0; r < 8; ++r) {
      float sv[4];
      float mx = -1e30f;
#pragma unroll
      for (int c = 0; c < 4; ++c) { sv[c] = s[c][r] * scale; mx = fmaxf(mx, sv[c]); }
#pragma unroll
      for (int off = 1; off < 16; off <<= 1) mx = fmaxf(mx, __shfl_xor(mx, off, 32));
      const float mn  = fmaxf(mrow[r], mx);
      const float fac = __expf(mrow[r] - mn);
      float rs = 0.0f;
#pragma unroll
      for (int c = 0; c < 4; ++c) { p[c][r] = __expf(sv[c] - mn); rs += p[c][r]; }
#pragma unroll
      for (int off = 1; off < 16; off <<= 1) rs += __shfl_xor(rs, off, 32);
      lrow[r] = lrow[r] * fac + rs;
      mrow[r] = mn;
      acc0[r] *= fac;
      acc1[r] *= fac;
    }

    // Re-stripe P: D layout -> row-major 16x64 LDS tile.
    __syncthreads();                    // previous iteration's reads are done
#pragma unroll
    for (int c = 0; c < 4; ++c) {
#pragma unroll
      for (int r = 0; r < 8; ++r) {
        const int m = r + half * 8;
        pTile[m * 64 + c * 16 + nn] = (_Float16)p[c][r];
      }
    }
    __syncthreads();
    v16h ap0 = load_frag((const _Float16*)pTile, 64);        // keys jb..jb+31
    v16h ap1 = load_frag((const _Float16*)pTile + 32, 64);   // keys jb+32..jb+63
    // V^T rows are e, cols are keys: contiguous B fragments.
    v16h bv00 = load_frag(vtb + jb, 1024);                   // e 0..15,  k 0..31
    v16h bv01 = load_frag(vtb + jb + 32, 1024);              // e 0..15,  k 32..63
    v16h bv10 = load_frag(vtb + (size_t)16 * 1024 + jb, 1024);
    v16h bv11 = load_frag(vtb + (size_t)16 * 1024 + jb + 32, 1024);
    acc0 = wmma_f16(ap0, bv00, acc0);
    acc0 = wmma_f16(ap1, bv01, acc0);
    acc1 = wmma_f16(ap0, bv10, acc1);
    acc1 = wmma_f16(ap1, bv11, acc1);
  }

  // Epilogue: normalize, re-stripe through LDS, contiguous b128 stores.
  __syncthreads();
#pragma unroll
  for (int r = 0; r < 8; ++r) {
    const int m = r + half * 8;
    const float inv = 1.0f / lrow[r];
    pTile[m * 32 + nn]      = (_Float16)(acc0[r] * inv);
    pTile[m * 32 + 16 + nn] = (_Float16)(acc1[r] * inv);
  }
  __syncthreads();
  store_tile_f16(attn + ((size_t)bg * 1024 + qt * 16) * E32, (const _Float16*)pTile);
}

// ---------------- Kernel 3: fused 3-layer grouped MLP ----------------
__global__ void __launch_bounds__(32)
mlp_kernel(const _Float16* __restrict__ v, const _Float16* __restrict__ attn,
           const float* __restrict__ w1,  const float* __restrict__ b1,
           const float* __restrict__ w10, const float* __restrict__ b10,
           const float* __restrict__ w11, const float* __restrict__ b11,
           float* __restrict__ out) {
  __shared__ alignas(16) _Float16 h1[16 * 32];
  __shared__ alignas(16) _Float16 h2[16 * 32];
  __shared__ alignas(16) float    ot[16 * 32];

  const int t = blockIdx.x;
  const int g = blockIdx.y;
  const int l = lane_id();
  const int rowbase = t * 16;
  const int b     = rowbase >> 10;
  const int nbase = rowbase & 1023;
  const int nn = l & 15, half = l >> 4;

  const size_t src = ((size_t)(b * GQ + g) * 1024 + nbase) * E32;
  v16h a_v  = load_frag(v + src, E32);
  v16h a_at = load_frag(attn + src, E32);

  // Layer 1: in 64 (v|attn), out 32.
#pragma unroll
  for (int cb = 0; cb < 2; ++cb) {
    const int ob = g * 32 + cb * 16;
    v16h bw0 = load_frag(w1 + (size_t)ob * 64, 64);
    v16h bw1 = load_frag(w1 + (size_t)ob * 64 + 32, 64);
    v8f acc = {};
    acc = wmma_f16(a_v, bw0, acc);
    acc = wmma_f16(a_at, bw1, acc);
    const float bn = b1[ob + nn];
#pragma unroll
    for (int r = 0; r < 8; ++r) {
      const int m = r + half * 8;
      const float z = acc[r] + bn;
      h1[m * 32 + cb * 16 + nn] = (_Float16)(z >= 0.0f ? z : 0.01f * z);
    }
  }
  __syncthreads();

  // Layer 2: in 32, out 32.
  v16h a_h1 = load_frag((const _Float16*)h1, 32);
#pragma unroll
  for (int cb = 0; cb < 2; ++cb) {
    const int ob = g * 32 + cb * 16;
    v16h bw = load_frag(w10 + (size_t)ob * 32, 32);
    v8f acc = {};
    acc = wmma_f16(a_h1, bw, acc);
    const float bn = b10[ob + nn];
#pragma unroll
    for (int r = 0; r < 8; ++r) {
      const int m = r + half * 8;
      const float z = acc[r] + bn;
      h2[m * 32 + cb * 16 + nn] = (_Float16)(z >= 0.0f ? z : 0.01f * z);
    }
  }
  __syncthreads();

  // Layer 3: in 32, out 32, f32 result.
  v16h a_h2 = load_frag((const _Float16*)h2, 32);
#pragma unroll
  for (int cb = 0; cb < 2; ++cb) {
    const int ob = g * 32 + cb * 16;
    v16h bw = load_frag(w11 + (size_t)ob * 32, 32);
    v8f acc = {};
    acc = wmma_f16(a_h2, bw, acc);
    const float bn = b11[ob + nn];
#pragma unroll
    for (int r = 0; r < 8; ++r) {
      const int m = r + half * 8;
      const float z = acc[r] + bn;
      ot[m * 32 + cb * 16 + nn] = (z >= 0.0f ? z : 0.01f * z);
    }
  }
  __syncthreads();
  // Re-striped f32 stores: 128-bit, 4 per lane; row pitch 64 floats, 32-col slice.
  float* dst = out + (size_t)rowbase * 64 + g * 32;
#pragma unroll
  for (int i = 0; i < 4; ++i) {
    const int idx = i * 32 + l;                 // float4 index (128 total)
    const int row = idx >> 3;
    const int c   = (idx & 7) * 4;
    *(uint4*)(dst + (size_t)row * 64 + c) = *(const uint4*)(ot + row * 32 + c);
  }
}

extern "C" void kernel_launch(void* const* d_in, const int* in_sizes, int n_in,
                              void* d_out, int out_size, void* d_ws, size_t ws_size,
                              hipStream_t stream) {
  const float* x     = (const float*)d_in[0];
  const float* w_qkv = (const float*)d_in[1];
  const float* b_qkv = (const float*)d_in[2];
  const float* w_h1  = (const float*)d_in[3];
  const float* b_h1  = (const float*)d_in[4];
  const float* w_h10 = (const float*)d_in[5];
  const float* b_h10 = (const float*)d_in[6];
  const float* w_h11 = (const float*)d_in[7];
  const float* b_h11 = (const float*)d_in[8];
  float* out = (float*)d_out;

  const size_t NEL = (size_t)32 * GQ * 1024 * E32;   // 2M f16 elements per buffer
  _Float16* ws   = (_Float16*)d_ws;
  _Float16* q    = ws;
  _Float16* k    = ws + NEL;
  _Float16* v    = ws + 2 * NEL;
  _Float16* vT   = ws + 3 * NEL;
  _Float16* attn = ws + 4 * NEL;

  qkv_kernel<<<dim3(2048, GQ), 32, 0, stream>>>(x, w_qkv, b_qkv, q, k, v, vT);
  attn_kernel<<<dim3(64, 64), 32, 0, stream>>>(q, k, vT, attn);
  mlp_kernel<<<dim3(2048, GQ), 32, 0, stream>>>(v, attn, w_h1, b_h1,
                                                w_h10, b_h10, w_h11, b_h11, out);
}
